// RNN_GRU_46325517255337
// MI455X (gfx1250) — matrix operand order repaired and checked by reading, not compiled, not run on hardware
//
#include <hip/hip_runtime.h>
#include <hip/hip_bf16.h>

// ---------------- types ----------------
typedef __attribute__((ext_vector_type(16))) __bf16 v16bf;
typedef __attribute__((ext_vector_type(8)))  __bf16 v8bf;
typedef __attribute__((ext_vector_type(8)))  float  v8f;

#define ZONEOUT 0.1f
#define GRU_U   512
#define GRU_B   32
#define GRU_T   1024
#define SCAN_NWG 32
#define SCAN_THREADS 192   // 6 waves: one (gate, mtile) WMMA job per wave

// ---------------- helpers ----------------
__device__ __forceinline__ v16bf cat8(v8bf lo, v8bf hi) {
    return __builtin_shufflevector(lo, hi, 0,1,2,3,4,5,6,7,8,9,10,11,12,13,14,15);
}

__device__ __forceinline__ void grid_sync(unsigned* cnt, unsigned* sense,
                                          unsigned phase, unsigned nwg) {
    __threadfence();
    __syncthreads();
    if (threadIdx.x == 0) {
        unsigned prev = __hip_atomic_fetch_add(cnt, 1u, __ATOMIC_ACQ_REL,
                                               __HIP_MEMORY_SCOPE_AGENT);
        if (prev == nwg - 1u) {
            __hip_atomic_store(cnt, 0u, __ATOMIC_RELAXED, __HIP_MEMORY_SCOPE_AGENT);
            __hip_atomic_store(sense, phase, __ATOMIC_RELEASE, __HIP_MEMORY_SCOPE_AGENT);
        } else {
            while (__hip_atomic_load(sense, __ATOMIC_ACQUIRE,
                                     __HIP_MEMORY_SCOPE_AGENT) != phase) {
                __builtin_amdgcn_s_sleep(1);
            }
        }
    }
    __syncthreads();
}

// ---------------- weight prep ----------------
// dst[c*R + r] = (bf16)src[r*C + c]   (src is [R,C] row-major; dst is [C,R])
__global__ void __launch_bounds__(256) transpose_to_bf16(
        const float* __restrict__ src, __bf16* __restrict__ dst, int R, int C) {
    int i = blockIdx.x * 256 + threadIdx.x;
    if (i >= R * C) return;
    int r = i / C, c = i % C;
    dst[(size_t)c * R + r] = (__bf16)src[i];
}

__global__ void __launch_bounds__(256) f32_to_bf16(
        const float* __restrict__ s, __bf16* __restrict__ d, int n) {
    int i = blockIdx.x * 256 + threadIdx.x;
    if (i < n) d[i] = (__bf16)s[i];
}

// d = bf16(o0 + a[0]*p1)
__global__ void __launch_bounds__(256) combine2_bf16(
        const float* __restrict__ o0, const float* __restrict__ p1,
        const float* __restrict__ a, __bf16* __restrict__ d, int n) {
    int i = blockIdx.x * 256 + threadIdx.x;
    if (i < n) d[i] = (__bf16)(o0[i] + a[0] * p1[i]);
}

// d = bf16(o0 + a[1]*p1 + a[2]*p2)
__global__ void __launch_bounds__(256) combine3_bf16(
        const float* __restrict__ o0, const float* __restrict__ p1,
        const float* __restrict__ p2, const float* __restrict__ a,
        __bf16* __restrict__ d, int n) {
    int i = blockIdx.x * 256 + threadIdx.x;
    if (i < n) d[i] = (__bf16)(o0[i] + a[1] * p1[i] + a[2] * p2[i]);
}

// broadcast h0[1,U] across batch into ping buffer 0 (f32 + bf16)
__global__ void __launch_bounds__(256) init_h(
        const float* __restrict__ h0, float* __restrict__ hf,
        __bf16* __restrict__ hb, int n, int U) {
    int i = blockIdx.x * 256 + threadIdx.x;
    if (i >= n) return;
    float v = h0[i % U];
    hf[i] = v;
    hb[i] = (__bf16)v;
}

// ---------------- bulk GEMM: C[M,N] = A[M,K](bf16) @ Bt[N,K](bf16)^T + bias ----------------
// 2x2 register-blocked: each wave computes a 32x32 block (4 WMMA tiles), reusing
// each A/B fragment twice. K multiple of 32, M,N multiples of 32.
// Tremap != 0 : output row r -> (r % Tremap)*(M/Tremap) + r/Tremap  (i.e. [B,T] -> [T,B])
__global__ void __launch_bounds__(256) gemm_bf16(
        const __bf16* __restrict__ A, const __bf16* __restrict__ Bt,
        const float* __restrict__ bias, float* __restrict__ C,
        int M, int N, int K, int Tremap) {
    const int wave = threadIdx.x >> 5;
    const int lane = threadIdx.x & 31;
    const int nblks = N >> 5;
    const long job   = (long)blockIdx.x * 8 + wave;
    const long total = (long)(M >> 5) * nblks;
    if (job >= total) return;                 // wave-uniform: EXEC stays all-ones
    const int mblk = (int)(job / nblks);
    const int nblk = (int)(job % nblks);

    // A: lane L holds row (mblk*32 [+16] + L%16); 16 contiguous K per lane
    const __bf16* a0 = A + (size_t)(mblk * 32 + (lane & 15)) * K + ((lane < 16) ? 0 : 16);
    const __bf16* a1 = a0 + (size_t)16 * K;
    // B: lane L holds column (nblk*32 [+16] + L%16); 16 contiguous K per lane
    const __bf16* b0 = Bt + (size_t)(nblk * 32 + (lane & 15)) * K + ((lane < 16) ? 0 : 16);
    const __bf16* b1 = b0 + (size_t)16 * K;

    v8f acc00 = {}, acc01 = {}, acc10 = {}, acc11 = {};
    for (int k0 = 0; k0 < K; k0 += 32) {
        v16bf av0 = cat8(*(const v8bf*)(a0 + k0), *(const v8bf*)(a0 + k0 + 8));
        v16bf av1 = cat8(*(const v8bf*)(a1 + k0), *(const v8bf*)(a1 + k0 + 8));
        v16bf bv0 = cat8(*(const v8bf*)(b0 + k0), *(const v8bf*)(b0 + k0 + 8));
        v16bf bv1 = cat8(*(const v8bf*)(b1 + k0), *(const v8bf*)(b1 + k0 + 8));
        __builtin_prefetch(a0 + k0 + 256, 0, 0);
        __builtin_prefetch(b0 + k0 + 256, 0, 0);
        acc00 = __builtin_amdgcn_wmma_f32_16x16x32_bf16(false, av0, false, bv0,
                                                        (short)0, acc00, false, false);
        acc01 = __builtin_amdgcn_wmma_f32_16x16x32_bf16(false, av0, false, bv1,
                                                        (short)0, acc01, false, false);
        acc10 = __builtin_amdgcn_wmma_f32_16x16x32_bf16(false, av1, false, bv0,
                                                        (short)0, acc10, false, false);
        acc11 = __builtin_amdgcn_wmma_f32_16x16x32_bf16(false, av1, false, bv1,
                                                        (short)0, acc11, false, false);
    }

    const int col0 = nblk * 32 + (lane & 15);
    const int col1 = col0 + 16;
    const float bv0s = bias ? bias[col0] : 0.0f;
    const float bv1s = bias ? bias[col1] : 0.0f;
    const int rbase = mblk * 32 + ((lane < 16) ? 0 : 8);
#pragma unroll
    for (int r = 0; r < 8; ++r) {
        int row0 = rbase + r;
        int row1 = row0 + 16;
        long o0 = Tremap ? ((long)(row0 % Tremap) * (M / Tremap) + row0 / Tremap) : (long)row0;
        long o1 = Tremap ? ((long)(row1 % Tremap) * (M / Tremap) + row1 / Tremap) : (long)row1;
        C[o0 * N + col0] = acc00[r] + bv0s;
        C[o0 * N + col1] = acc01[r] + bv1s;
        C[o1 * N + col0] = acc10[r] + bv0s;
        C[o1 * N + col1] = acc11[r] + bv1s;
    }
}

// ---------------- persistent GRU scan ----------------
// xw:  [T, B, 3U] f32 (b_in already added)
// Rt:  [3U, U] bf16 (transposed recurrent: Rt[n][k] = R[k][n])
// brec:[3U] f32
// hf:  [2, B, U] f32 ping-pong carried state; hb: bf16 mirror (WMMA input)
// ys:  [B, T, U] f32 pre-zoneout outputs
// Recurrent weights are time-invariant: each wave pins its 16 B-fragments
// (128 VGPRs) in registers for all T steps; inner loop = 2 global b128 loads
// of h + 1 WMMA per k-step. LDS only used for the 6KB gate-exchange tile.
__global__ void __launch_bounds__(SCAN_THREADS) gru_scan(
        const float* __restrict__ xw, const __bf16* __restrict__ Rt,
        const float* __restrict__ brec,
        float* __restrict__ hf, __bf16* __restrict__ hb,
        float* __restrict__ ys,
        unsigned* __restrict__ bar_cnt, unsigned* __restrict__ bar_sense,
        int T) {
    __shared__ float lds_m[3 * GRU_B * 16]; // m exchange tile (6 KB)

    const int tid  = threadIdx.x;
    const int lane = tid & 31;
    const int wave = tid >> 5;
    const int u0   = blockIdx.x * 16;        // this WG's hidden-unit slice

    const int gate  = wave % 3;
    const int mtile = wave / 3;              // batch 0..15 or 16..31
    const int arow  = mtile * 16 + (lane & 15);
    const int aoff  = (lane < 16) ? 0 : 16;

    // ---- pin this wave's recurrent-weight fragments in VGPRs (once) ----
    const __bf16* rtrow = Rt + (size_t)(gate * GRU_U + u0 + (lane & 15)) * GRU_U
                             + ((lane < 16) ? 0 : 16);
    v16bf bw[16];
#pragma unroll
    for (int k = 0; k < 16; ++k)
        bw[k] = cat8(*(const v8bf*)(rtrow + 32 * k), *(const v8bf*)(rtrow + 32 * k + 8));

    for (int t = 0; t < T; ++t) {
        const __bf16* hcur = hb + (size_t)(t & 1) * GRU_B * GRU_U;
        const float*  xwt  = xw + (size_t)t * GRU_B * 3 * GRU_U;

        // ---- m = h @ R  (this wave's 16x16 tile, K=512) ----
        v8f acc = {};
        const __bf16* ap = hcur + (size_t)arow * GRU_U + aoff;
#pragma unroll
        for (int k = 0; k < 16; ++k) {
            v16bf av = cat8(*(const v8bf*)(ap + 32 * k), *(const v8bf*)(ap + 32 * k + 8));
            acc = __builtin_amdgcn_wmma_f32_16x16x32_bf16(
                    false, av, false, bw[k], (short)0, acc, false, false);
        }
        // scatter C tile to LDS: lane<16 rows +0..7, lane>=16 rows +8..15
        {
            int mloc = mtile * 16 + ((lane < 16) ? 0 : 8);
            int col  = lane & 15;
#pragma unroll
            for (int r = 0; r < 8; ++r)
                lds_m[gate * (GRU_B * 16) + (mloc + r) * 16 + col] = acc[r];
        }
        __syncthreads();

        // ---- fused gates + zoneout ----
        const int nxt = (t + 1) & 1;
        const float* hfc = hf + (size_t)(t & 1) * GRU_B * GRU_U;
        float*  hfn = hf + (size_t)nxt * GRU_B * GRU_U;
        __bf16* hbn = hb + (size_t)nxt * GRU_B * GRU_U;
        for (int i = tid; i < GRU_B * 16; i += SCAN_THREADS) {
            int b = i >> 4, c = i & 15;
            int u = u0 + c;
            float xz = xwt[b * 1536 + u];
            float xr = xwt[b * 1536 + 512 + u];
            float xh = xwt[b * 1536 + 1024 + u];
            // prefetch next step's xw slab (predictable 192KB/step L2 stream)
            __builtin_prefetch(xwt + GRU_B * 3 * GRU_U + b * 1536 + u, 0, 0);
            float mz = lds_m[0 * 512 + b * 16 + c] + brec[u];
            float mr = lds_m[1 * 512 + b * 16 + c] + brec[512 + u];
            float mh = lds_m[2 * 512 + b * 16 + c] + brec[1024 + u];
            float hp = hfc[b * GRU_U + u];
            float z  = 1.0f / (1.0f + __expf(-(xz + mz)));
            float r  = 1.0f / (1.0f + __expf(-(xr + mr)));
            float hh = tanhf(xh + r * mh);
            float hn = z * hp + (1.0f - z) * hh;
            ys[((size_t)b * T + t) * GRU_U + u] = hn;
            float hc = hn + ZONEOUT * (hp - hn);
            hfn[b * GRU_U + u] = hc;
            hbn[b * GRU_U + u] = (__bf16)hc;
        }
        grid_sync(bar_cnt, bar_sense, (unsigned)(t + 1), SCAN_NWG);
    }
}

// ---------------- host launch ----------------
static inline size_t al256(size_t x) { return (x + 255) & ~(size_t)255; }

extern "C" void kernel_launch(void* const* d_in, const int* in_sizes, int n_in,
                              void* d_out, int out_size, void* d_ws, size_t ws_size,
                              hipStream_t stream) {
    (void)in_sizes; (void)n_in; (void)out_size; (void)ws_size;
    const float* x   = (const float*)d_in[0];   // [32,1024,256]
    const float* k0  = (const float*)d_in[1];   // [256,1536]
    const float* r0  = (const float*)d_in[2];   // [512,1536]
    const float* b0  = (const float*)d_in[3];   // [2,1536]
    const float* ks  = (const float*)d_in[4];   // [512,1536]
    const float* rs  = (const float*)d_in[5];   // [512,1536]
    const float* bs  = (const float*)d_in[6];   // [2,1536]
    const float* h00 = (const float*)d_in[7];   // [1,512]
    const float* h01 = (const float*)d_in[8];
    const float* h02 = (const float*)d_in[9];
    const float* a   = (const float*)d_in[10];  // [3]
    const float* wd  = (const float*)d_in[11];  // [512,256]
    const float* bd  = (const float*)d_in[12];  // [256]
    float* out = (float*)d_out;                 // [32,1024,256]

    const int B = GRU_B, T = GRU_T, D = 256, U = GRU_U;
    const size_t BT = (size_t)B * T;

    // workspace layout
    char* ws = (char*)d_ws;
    size_t cur = 0;
    auto take = [&](size_t bytes) { char* p = ws + cur; cur = al256(cur + bytes); return p; };
    float*  xw       = (float*) take(BT * 3 * U * sizeof(float));   // [T,B,3U]
    __bf16* x_bf     = (__bf16*)take((size_t)BT * D * 2);
    __bf16* act_bf   = (__bf16*)take(BT * U * 2);
    float*  out0     = (float*) take(BT * U * 4);
    float*  pred1    = (float*) take(BT * U * 4);
    float*  pred2    = (float*) take(BT * U * 4);
    __bf16* k0t      = (__bf16*)take((size_t)3 * U * D * 2);        // [1536,256]
    __bf16* r0t      = (__bf16*)take((size_t)3 * U * U * 2);        // [1536,512]
    __bf16* kst      = (__bf16*)take((size_t)3 * U * U * 2);
    __bf16* rst      = (__bf16*)take((size_t)3 * U * U * 2);
    __bf16* wdt      = (__bf16*)take((size_t)D * U * 2);            // [256,512]
    float*  hf       = (float*) take((size_t)2 * B * U * 4);
    __bf16* hbuf     = (__bf16*)take((size_t)2 * B * U * 2);
    unsigned* bars   = (unsigned*)take(3 * 256);

    hipMemsetAsync(bars, 0, 3 * 256, stream);

    // weight prep
    transpose_to_bf16<<<(D * 3 * U + 255) / 256, 256, 0, stream>>>(k0, k0t, D, 3 * U);
    transpose_to_bf16<<<(U * 3 * U + 255) / 256, 256, 0, stream>>>(r0, r0t, U, 3 * U);
    transpose_to_bf16<<<(U * 3 * U + 255) / 256, 256, 0, stream>>>(ks, kst, U, 3 * U);
    transpose_to_bf16<<<(U * 3 * U + 255) / 256, 256, 0, stream>>>(rs, rst, U, 3 * U);
    transpose_to_bf16<<<(U * D + 255) / 256, 256, 0, stream>>>(wd, wdt, U, D);
    f32_to_bf16<<<(int)((BT * D + 255) / 256), 256, 0, stream>>>(x, x_bf, (int)(BT * D));

    const int gemm_xw_blocks  = (int)(((BT / 32) * ((3 * U) / 32)) / 8);   // 6144
    const int gemm_out_blocks = (int)(((BT / 32) * (D / 32)) / 8);         // 1024

    // ---- layer 0 ----
    gemm_bf16<<<gemm_xw_blocks, 256, 0, stream>>>(x_bf, k0t, b0, xw,
                                                  (int)BT, 3 * U, D, T);
    init_h<<<(B * U + 255) / 256, 256, 0, stream>>>(h00, hf, hbuf, B * U, U);
    gru_scan<<<SCAN_NWG, SCAN_THREADS, 0, stream>>>(xw, r0t, b0 + 3 * U,
                                                    hf, hbuf, out0,
                                                    bars + 0, bars + 16, T);

    // ---- layer 1 (pred1) ----
    f32_to_bf16<<<(int)((BT * U + 255) / 256), 256, 0, stream>>>(out0, act_bf, (int)(BT * U));
    gemm_bf16<<<gemm_xw_blocks, 256, 0, stream>>>(act_bf, kst, bs, xw,
                                                  (int)BT, 3 * U, U, T);
    init_h<<<(B * U + 255) / 256, 256, 0, stream>>>(h01, hf, hbuf, B * U, U);
    gru_scan<<<SCAN_NWG, SCAN_THREADS, 0, stream>>>(xw, rst, bs + 3 * U,
                                                    hf, hbuf, pred1,
                                                    bars + 64, bars + 80, T);

    // ---- layer 2 (pred2), input = out0 + a[0]*pred1 ----
    combine2_bf16<<<(int)((BT * U + 255) / 256), 256, 0, stream>>>(out0, pred1, a,
                                                                   act_bf, (int)(BT * U));
    gemm_bf16<<<gemm_xw_blocks, 256, 0, stream>>>(act_bf, kst, bs, xw,
                                                  (int)BT, 3 * U, U, T);
    init_h<<<(B * U + 255) / 256, 256, 0, stream>>>(h02, hf, hbuf, B * U, U);
    gru_scan<<<SCAN_NWG, SCAN_THREADS, 0, stream>>>(xw, rst, bs + 3 * U,
                                                    hf, hbuf, pred2,
                                                    bars + 128, bars + 144, T);

    // ---- head: (out0 + a[1]*pred1 + a[2]*pred2) @ wd + bd ----
    combine3_bf16<<<(int)((BT * U + 255) / 256), 256, 0, stream>>>(out0, pred1, pred2, a,
                                                                   act_bf, (int)(BT * U));
    gemm_bf16<<<gemm_out_blocks, 256, 0, stream>>>(act_bf, wdt, bd, out,
                                                   (int)BT, D, U, 0);
}
